// TritonFuseAll_14267881357585
// MI455X (gfx1250) — matrix operand order repaired and checked by reading, not compiled, not run on hardware
//
#include <hip/hip_runtime.h>
#include <hip/hip_bf16.h>
#include <math.h>

// Fused FC -> log_softmax -> NLL(mean) for N=4096, D=1024, V=50257 on gfx1250.
// bf16 down-convert pre-pass, then v_wmma_f32_16x16x32_bf16 GEMM tiles with a
// fully unrolled, register double-buffered K loop (all loads use immediate
// offsets off loop-invariant base pointers), in-block softmax partials, then
// a two-stage logsumexp/mean reduction.
// Workspace requirement: ~118 MB (bf16 copies of x and W + partials).

typedef __attribute__((ext_vector_type(16))) __bf16 v16bf;
typedef __attribute__((ext_vector_type(8)))  float  v8f;

#define N_ROWS 4096
#define DDIM   1024
#define VSIZE  50257
#define BM     32            // rows per block
#define BN     256           // vocab columns per block
#define NCHUNK ((VSIZE + BN - 1) / BN)   // 197

// ---------------------------------------------------------------------------
// Kernel 1: f32 -> bf16 conversion (grid-stride, float4 in / 4x bf16 out)
// ---------------------------------------------------------------------------
__global__ void cvt_f32_to_bf16(const float* __restrict__ in,
                                __bf16* __restrict__ out, int n4) {
  int i = blockIdx.x * blockDim.x + threadIdx.x;
  int stride = gridDim.x * blockDim.x;
  for (; i < n4; i += stride) {
    float4 f = ((const float4*)in)[i];
    union { __bf16 h[4]; uint2 q; } o;
    o.h[0] = (__bf16)f.x; o.h[1] = (__bf16)f.y;
    o.h[2] = (__bf16)f.z; o.h[3] = (__bf16)f.w;
    ((uint2*)out)[i] = o.q;
  }
}

// ---------------------------------------------------------------------------
// Fragment helpers.
// A fragment (16-bit A 16x32): lane m=L%16, g=L/16, elem e -> K=e+16g.
//   The 16g term is folded into the base pointer; remaining K-pair offsets are
//   the compile-time constants {0,2,4,6,8,10,12,14}.
// B fragment (16-bit B 32x16): lane n=L%16, g=L/16, elem e -> K=e+16g
//   (16 contiguous K == 32 contiguous bytes of one W row; 16g folded in base).
// C/D: lane n=L%16, VGPR r -> row m = r + 8*(L/16).
// ---------------------------------------------------------------------------
union AFrag { uint u[8]; v16bf v; };

__device__ __forceinline__ void load_a(const __bf16* __restrict__ xg,
                                       int k0, AFrag& af) {
#pragma unroll
  for (int j = 0; j < 8; ++j) {
    const int kb = 2 * j;   // compile-time constant
    af.u[j] = *(const uint*)(xg + k0 + kb);
  }
}

__device__ __forceinline__ void load_b(const __bf16* const (&wg)[4],
                                       int k0, v16bf (&b)[4]) {
#pragma unroll
  for (int a = 0; a < 4; ++a)
    b[a] = *(const v16bf*)(wg[a] + k0);
}

__device__ __forceinline__ void mma4(const AFrag& af, const v16bf (&b)[4],
                                     v8f (&acc)[4]) {
#pragma unroll
  for (int a = 0; a < 4; ++a)
    acc[a] = __builtin_amdgcn_wmma_f32_16x16x32_bf16(
        false, af.v, false, b[a], (short)0, acc[a], false, false);
}

// ---------------------------------------------------------------------------
// Kernel 2: WMMA GEMM tile + in-block softmax partials.
// Block: 256 threads = 8 wave32. Wave (wm,wn): wm in {0,1} picks 16-row band,
// wn in {0..3} picks a 64-column slab -> 4 accumulators of 16x16.
// ---------------------------------------------------------------------------
__global__ __launch_bounds__(256, 1)
void ce_partial_kernel(const __bf16* __restrict__ xb,
                       const __bf16* __restrict__ wb,
                       const int*    __restrict__ tgt,
                       float* __restrict__ pm,   // [NCHUNK][N_ROWS] row max
                       float* __restrict__ ps,   // [NCHUNK][N_ROWS] sum exp
                       float* __restrict__ tl) { // [N_ROWS] target logit
  __shared__ float ldsM[BM][4];
  __shared__ float ldsS[BM][4];

  const int tid  = threadIdx.x;
  const int wave = tid >> 5;
  const int lane = tid & 31;
  const int wm   = wave & 1;      // 16-row band within 32-row tile
  const int wn   = wave >> 1;     // 64-col slab within 256-col tile
  const int g    = lane >> 4;
  const int n16  = lane & 15;
  const int rowBase = blockIdx.x * BM;
  const int v0      = blockIdx.y * BN;

  v8f acc[4] = {};

  // loop-invariant per-lane base pointers (lane-group terms folded in)
  const __bf16* xg = xb + (size_t)(rowBase + wm * 16 + n16) * DDIM + 16 * g;
  const __bf16* wg[4];
#pragma unroll
  for (int a = 0; a < 4; ++a) {
    int v = v0 + wn * 64 + a * 16 + n16;
    if (v >= VSIZE) v = VSIZE - 1;          // clamp; masked in reduction
    wg[a] = wb + (size_t)v * DDIM + 16 * g;
  }

  // ---- fully unrolled, register double-buffered K loop ----
  AFrag a0, a1;
  v16bf b0[4], b1[4];
  load_a(xg, 0, a0);
  load_b(wg, 0, b0);

#pragma unroll
  for (int kk = 0; kk < DDIM / 64; ++kk) {
    const int k0 = kk * 64;
    const int k1 = k0 + 32;
    const int k2 = (kk + 1 < DDIM / 64) ? k0 + 64 : 0;  // redundant, no OOB

    // pull the upcoming W slab toward GL2 (global_prefetch_b8; no counters)
    __builtin_prefetch(wg[0] + k0 + 256, 0, 1);

    load_a(xg, k1, a1);               // prefetch k+32 into the idle buffer
    load_b(wg, k1, b1);
    mma4(a0, b0, acc);                // compute step k

    load_a(xg, k2, a0);               // prefetch k+64
    load_b(wg, k2, b0);
    mma4(a1, b1, acc);                // compute step k+32
  }

  // ---- per-row max over this block's 256 columns ----
#pragma unroll
  for (int r = 0; r < 8; ++r) {
    float mx = -INFINITY;
#pragma unroll
    for (int a = 0; a < 4; ++a) {
      int col = v0 + wn * 64 + a * 16 + n16;
      if (col < VSIZE) mx = fmaxf(mx, acc[a][r]);
    }
#pragma unroll
    for (int off = 1; off < 16; off <<= 1)
      mx = fmaxf(mx, __shfl_xor(mx, off, 32));
    if (n16 == 0) ldsM[wm * 16 + r + 8 * g][wn] = mx;
  }
  __syncthreads();

  // ---- per-row sum of exp(logit - rowmax), plus target-logit capture ----
#pragma unroll
  for (int r = 0; r < 8; ++r) {
    int rowl = wm * 16 + r + 8 * g;
    float M = fmaxf(fmaxf(ldsM[rowl][0], ldsM[rowl][1]),
                    fmaxf(ldsM[rowl][2], ldsM[rowl][3]));
    float s = 0.f;
#pragma unroll
    for (int a = 0; a < 4; ++a) {
      int col = v0 + wn * 64 + a * 16 + n16;
      if (col < VSIZE) s += __expf(acc[a][r] - M);
    }
#pragma unroll
    for (int off = 1; off < 16; off <<= 1)
      s += __shfl_xor(s, off, 32);
    if (n16 == 0) ldsS[rowl][wn] = s;

    int grow = rowBase + rowl;
    int t = tgt[grow];
#pragma unroll
    for (int a = 0; a < 4; ++a) {
      int col = v0 + wn * 64 + a * 16 + n16;
      if (col == t) tl[grow] = acc[a][r];   // exactly one writer per row
    }
  }
  __syncthreads();

  if (tid < BM) {
    float M = fmaxf(fmaxf(ldsM[tid][0], ldsM[tid][1]),
                    fmaxf(ldsM[tid][2], ldsM[tid][3]));
    float S = ldsS[tid][0] + ldsS[tid][1] + ldsS[tid][2] + ldsS[tid][3];
    size_t idx = (size_t)blockIdx.y * N_ROWS + (size_t)(rowBase + tid);
    pm[idx] = M;
    ps[idx] = S;
  }
}

// ---------------------------------------------------------------------------
// Kernel 3: online logsumexp merge over chunks, one thread per row
// ---------------------------------------------------------------------------
__global__ void ce_combine_kernel(const float* __restrict__ pm,
                                  const float* __restrict__ ps,
                                  const float* __restrict__ tl,
                                  float* __restrict__ nll) {
  int row = blockIdx.x * blockDim.x + threadIdx.x;
  if (row >= N_ROWS) return;
  float M = -INFINITY, S = 0.f;
  for (int c = 0; c < NCHUNK; ++c) {
    float m = pm[(size_t)c * N_ROWS + row];
    float s = ps[(size_t)c * N_ROWS + row];
    if (m > M) { S = S * __expf(M - m) + s; M = m; }
    else       { S += s * __expf(m - M); }
  }
  nll[row] = M + __logf(S) - tl[row];
}

// ---------------------------------------------------------------------------
// Kernel 4: mean over 4096 rows -> d_out[0]
// ---------------------------------------------------------------------------
__global__ void ce_mean_kernel(const float* __restrict__ nll,
                               float* __restrict__ out) {
  __shared__ float red[256];
  float s = 0.f;
  for (int i = threadIdx.x; i < N_ROWS; i += 256) s += nll[i];
  red[threadIdx.x] = s;
  __syncthreads();
  for (int off = 128; off > 0; off >>= 1) {
    if (threadIdx.x < off) red[threadIdx.x] += red[threadIdx.x + off];
    __syncthreads();
  }
  if (threadIdx.x == 0) out[0] = red[0] / (float)N_ROWS;
}

// ---------------------------------------------------------------------------
extern "C" void kernel_launch(void* const* d_in, const int* in_sizes, int n_in,
                              void* d_out, int out_size, void* d_ws, size_t ws_size,
                              hipStream_t stream) {
  const float* x   = (const float*)d_in[0];   // [4096, 1024] f32
  const float* w   = (const float*)d_in[1];   // [50257, 1024] f32
  const int*   tgt = (const int*)d_in[2];     // [4096] int
  float* out = (float*)d_out;                 // scalar loss

  // workspace carve-out (256B aligned slabs)
  char* ws = (char*)d_ws;
  size_t off = 0;
  auto carve = [&](size_t bytes) -> void* {
    void* p = ws + off;
    off += (bytes + 255) & ~(size_t)255;
    return p;
  };
  __bf16* xb  = (__bf16*)carve((size_t)N_ROWS * DDIM * sizeof(__bf16));
  __bf16* wb  = (__bf16*)carve((size_t)VSIZE  * DDIM * sizeof(__bf16));
  float*  pm  = (float*) carve((size_t)NCHUNK * N_ROWS * sizeof(float));
  float*  ps  = (float*) carve((size_t)NCHUNK * N_ROWS * sizeof(float));
  float*  tl  = (float*) carve((size_t)N_ROWS * sizeof(float));
  float*  nll = (float*) carve((size_t)N_ROWS * sizeof(float));

  // 1) down-convert inputs to bf16 (bandwidth-bound, ~334 MB traffic)
  cvt_f32_to_bf16<<<2048, 256, 0, stream>>>(x, xb, (N_ROWS * DDIM) / 4);
  cvt_f32_to_bf16<<<8192, 256, 0, stream>>>(w, wb, (VSIZE * DDIM) / 4);

  // 2) WMMA GEMM + per-tile softmax partials.
  //    grid.x = row tiles (fast-varying -> consecutive blocks share the same
  //    512 KB W slab, captured by the 192 MB L2), grid.y = vocab chunks.
  dim3 grid(N_ROWS / BM, NCHUNK);
  ce_partial_kernel<<<grid, 256, 0, stream>>>(xb, wb, tgt, pm, ps, tl);

  // 3) logsumexp merge per row
  ce_combine_kernel<<<N_ROWS / 256, 256, 0, stream>>>(pm, ps, tl, nll);

  // 4) mean -> scalar
  ce_mean_kernel<<<1, 256, 0, stream>>>(nll, out);
}
